// Net_61881888800841
// MI455X (gfx1250) — compile-verified, matrix operand order checked by reading, not run on hardware
//
#include <hip/hip_runtime.h>

typedef __attribute__((ext_vector_type(16))) _Float16 v16h;
typedef __attribute__((ext_vector_type(16))) short    v16s;
typedef __attribute__((ext_vector_type(8)))  float    v8f;

#define NC_EPS 1e-20f
#define ZPAD_BASE 800   // 16 zero slots at [800..815] in each LDS array

__device__ __forceinline__ float softplusf(float x) {
    return (x > 20.f) ? x : log1pf(expf(x));
}

// ---------------------------------------------------------------------------
// Weight packing: softplus(w_raw) -> WMMA B-fragment panels, per-out-channel
// weight sums, and the K -> LDS-offset table used to build A fragments.
//
// B fragment (32x16 f16, K x N), wave32: lane L holds 16 halves.
//   element i: v=i>>1, h=i&1
//   K = (v>>2)*16 + (L>>4)*8 + (v&3)*2 + h ,  N = L & 15
// A fragment (16x32 f16, M x K), wave32: lane L holds M=L&15; same K decode.
// Offset table entry (per chunk, lane, element): LDS offset of (ci,kh,kw)
// window element (final address = entry + M); K-padding points at zero slots.
// w_raw layout (O, I, kh, kw): flat index n*Ktot + kg, kg = ci*k*k + kh*k + kw
// ---------------------------------------------------------------------------
__global__ __launch_bounds__(32)
void pack_weights_kernel(const float* __restrict__ w_raw,
                         _Float16* __restrict__ panel,
                         short* __restrict__ offtab,
                         float* __restrict__ wsum,
                         int Cout, int Cin, int ksz)
{
    const int lane  = threadIdx.x;      // 0..31
    const int chunk = blockIdx.x;
    const int kk    = ksz * ksz;
    const int Ktot  = Cin * kk;
    const int TW    = 16 + ksz - 1;
    const int n     = lane & 15;
    const int khalf = lane >> 4;
    _Float16* dstW = panel  + (size_t)chunk * 512 + (size_t)lane * 16;
    short*    dstO = offtab + (size_t)chunk * 512 + (size_t)lane * 16;
    for (int i = 0; i < 16; ++i) {
        int v  = i >> 1, h = i & 1;
        int kl = ((v >> 2) << 4) + (khalf << 3) + ((v & 3) << 1) + h;
        int kg = chunk * 32 + kl;
        float val = 0.f;
        short o   = (short)ZPAD_BASE;   // points at zeroed LDS slots
        if (kg < Ktot) {
            int ci  = kg / kk;
            int rem = kg - ci * kk;
            int kh  = rem / ksz;
            int kw  = rem - kh * ksz;
            o = (short)((ci * ksz + kh) * TW + kw);
            if (n < Cout) val = softplusf(w_raw[n * Ktot + kg]);
        }
        dstW[i] = (_Float16)val;
        dstO[i] = o;
    }
    if (chunk == 0 && lane < Cout) {
        float s = 0.f;
        for (int j = 0; j < Ktot; ++j) s += softplusf(w_raw[lane * Ktot + j]);
        wsum[lane] = s;
    }
}

// ---------------------------------------------------------------------------
// Normalized convolution via implicit-GEMM WMMA.
// One wave (32 threads) per (batch, out-row, 16-pixel tile).
//   M = 16 output pixels, N = 16 out channels (Cout real), K = Cin*k*k in
//   chunks of 32 (f16).  Two accumulators share B: nominator (x*c) + denom (c).
// Concat inputs: channels [0,chA) from (xA,cA), rest from (xB,cB).
// Hot loop is division/branch free: offsets come from the precomputed table.
// ---------------------------------------------------------------------------
__global__ __launch_bounds__(32)
void nconv_wmma_kernel(const float* __restrict__ xA, const float* __restrict__ cA, int chA,
                       const float* __restrict__ xB, const float* __restrict__ cB,
                       float* __restrict__ x_out, float* __restrict__ c_out,
                       const _Float16* __restrict__ panel,
                       const short* __restrict__ offtab, int nChunks,
                       const float* __restrict__ wsum, const float* __restrict__ bias,
                       int Cin, int Hin, int Win, int ksz, int pad,
                       int Hout, int Wout, int Cout)
{
    __shared__ float sXC[ZPAD_BASE + 16];   // max window 8*5*20=800 + 16 zero slots
    __shared__ float sC [ZPAD_BASE + 16];

    const int lane = threadIdx.x;
    const int x0   = blockIdx.x * 16;
    const int oy   = blockIdx.y;
    const int b    = blockIdx.z;
    const int TW   = 16 + ksz - 1;

    // Zero-pad slots used by the K-padding entries of the offset table.
    if (lane < 16) { sXC[ZPAD_BASE + lane] = 0.f; sC[ZPAD_BASE + lane] = 0.f; }

    // Stage input window (zero padded): x*c and c.  Division free.
    {
        const int col = lane;
        const int gx  = x0 + col - pad;
        const bool colOK = (col < TW) && (gx >= 0) && (gx < Win);
        for (int ci = 0; ci < Cin; ++ci) {
            const float* xs; const float* cs; int cc, chS;
            if (ci < chA) { xs = xA; cs = cA; cc = ci;       chS = chA; }
            else          { xs = xB; cs = cB; cc = ci - chA; chS = Cin - chA; }
            const size_t rowBase = ((size_t)b * chS + cc) * Hin;
            for (int kh = 0; kh < ksz; ++kh) {
                int gy = oy + kh - pad;
                float xv = 0.f, cv = 0.f;
                if (colOK && gy >= 0 && gy < Hin) {
                    size_t off = (rowBase + gy) * (size_t)Win + gx;
                    xv = xs[off]; cv = cs[off];
                }
                if (col < TW) {
                    int li = (ci * ksz + kh) * TW + col;
                    sXC[li] = xv * cv;
                    sC[li]  = cv;
                }
            }
        }
    }
    __syncthreads();

    v8f accN = {0.f,0.f,0.f,0.f,0.f,0.f,0.f,0.f};
    v8f accD = {0.f,0.f,0.f,0.f,0.f,0.f,0.f,0.f};
    const int m0    = lane & 15;   // A-matrix row for this lane
    const int khalf = lane >> 4;

    for (int ch = 0; ch < nChunks; ++ch) {
        v16h bfrag = *(const v16h*)(panel  + (size_t)ch * 512 + (size_t)lane * 16);
        v16s ot    = *(const v16s*)(offtab + (size_t)ch * 512 + (size_t)lane * 16);
        v16h aN, aD;
        #pragma unroll
        for (int i = 0; i < 16; ++i) {
            int li = (int)ot[i] + m0;      // zero-pad entries land in [800..815]
            aN[i] = (_Float16)sXC[li];
            aD[i] = (_Float16)sC[li];
        }
        accN = __builtin_amdgcn_wmma_f32_16x16x32_f16(false, aN, false, bfrag,
                                                      (short)0, accN, false, false);
        accD = __builtin_amdgcn_wmma_f32_16x16x32_f16(false, aD, false, bfrag,
                                                      (short)0, accD, false, false);
    }

    // Epilogue: C/D layout -> lane has N = lane&15, rows M = r + 8*(lane>>4)
    const int n = lane & 15;
    if (n < Cout) {
        float bv = bias[n];
        float ws = wsum[n];
        #pragma unroll
        for (int r = 0; r < 8; ++r) {
            int m  = r + (khalf << 3);
            int gx = x0 + m;
            if (gx < Wout) {
                size_t off = (((size_t)b * Cout + n) * Hout + oy) * (size_t)Wout + gx;
                float den = accD[r];
                x_out[off] = accN[r] / (den + NC_EPS) + bv;
                if (c_out) c_out[off] = den / ws;
            }
        }
    }
}

// ---------------------------------------------------------------------------
__global__ void thresh_kernel(const float* __restrict__ S, float* __restrict__ c0, int total)
{
    int i = blockIdx.x * blockDim.x + threadIdx.x;
    if (i < total) c0[i] = (S[i] > 0.01f) ? 1.f : 0.f;
}

// max-pool 2x2 on c, gather x at first-argmax (order (0,0),(0,1),(1,0),(1,1)); c scaled 0.25
__global__ void pool2_kernel(const float* __restrict__ x, const float* __restrict__ c,
                             float* __restrict__ xo, float* __restrict__ co,
                             int C, int h2, int w2, int Hin, int Win, int total)
{
    int i = blockIdx.x * blockDim.x + threadIdx.x;
    if (i >= total) return;
    int pw = i % w2; int t = i / w2;
    int ph = t % h2; t /= h2;
    int cc = t % C;  int b = t / C;
    size_t base = (((size_t)b * C + cc) * Hin + ph * 2) * (size_t)Win + pw * 2;
    float cv0 = c[base],       cv1 = c[base + 1];
    float cv2 = c[base + Win], cv3 = c[base + Win + 1];
    float xv0 = x[base],       xv1 = x[base + 1];
    float xv2 = x[base + Win], xv3 = x[base + Win + 1];
    float bc = cv0, bx = xv0;
    if (cv1 > bc) { bc = cv1; bx = xv1; }
    if (cv2 > bc) { bc = cv2; bx = xv2; }
    if (cv3 > bc) { bc = cv3; bx = xv3; }
    size_t o = (((size_t)b * C + cc) * h2 + ph) * (size_t)w2 + pw;
    xo[o] = bx;
    co[o] = bc * 0.25f;
}

// nearest-neighbor 2x upsample
__global__ void up2_kernel(const float* __restrict__ in, float* __restrict__ out,
                           int C, int hi, int wi, int total)
{
    int i = blockIdx.x * blockDim.x + threadIdx.x;
    if (i >= total) return;
    int wo = 2 * wi, ho = 2 * hi;
    int xx = i % wo; int t = i / wo;
    int yy = t % ho; t /= ho;
    int cc = t % C;  int b = t / C;
    out[i] = in[(((size_t)b * C + cc) * hi + (yy >> 1)) * (size_t)wi + (xx >> 1)];
}

// adaptive average resize (hi,wi)->(Ho,Wo), single channel per batch
__global__ void resize_kernel(const float* __restrict__ xin, float* __restrict__ out,
                              int hi, int wi, int Ho, int Wo, int total)
{
    int i = blockIdx.x * blockDim.x + threadIdx.x;
    if (i >= total) return;
    int ox = i % Wo; int t = i / Wo;
    int oy = t % Ho; int b = t / Ho;
    int sy = (oy * hi) / Ho, ey = ((oy + 1) * hi + Ho - 1) / Ho;
    int sx = (ox * wi) / Wo, ex = ((ox + 1) * wi + Wo - 1) / Wo;
    float s = 0.f;
    for (int yy = sy; yy < ey; ++yy)
        for (int xx = sx; xx < ex; ++xx)
            s += xin[((size_t)b * hi + yy) * (size_t)wi + xx];
    out[i] = s / (float)((ey - sy) * (ex - sx));
}

// ---------------------------------------------------------------------------
extern "C" void kernel_launch(void* const* d_in, const int* in_sizes, int n_in,
                              void* d_out, int out_size, void* d_ws, size_t ws_size,
                              hipStream_t stream)
{
    (void)in_sizes; (void)n_in; (void)out_size; (void)ws_size;
    const int B = 8, H = 480, W = 640;

    const float* S = (const float*)d_in[1];
    const float* Wt[8] = { (const float*)d_in[3],  (const float*)d_in[5],
                           (const float*)d_in[7],  (const float*)d_in[9],
                           (const float*)d_in[11], (const float*)d_in[13],
                           (const float*)d_in[15], (const float*)d_in[17] };
    const float* Bs[8] = { (const float*)d_in[4],  (const float*)d_in[6],
                           (const float*)d_in[8],  (const float*)d_in[10],
                           (const float*)d_in[12], (const float*)d_in[14],
                           (const float*)d_in[16], (const float*)d_in[18] };

    // layer meta: w1 w2 w3 w4 w5 w6 w65 w7
    const int mCout[8] = {4,4,4,4,4,4,4,1};
    const int mCin [8] = {1,4,4,8,8,8,4,4};
    const int mK   [8] = {5,5,5,3,3,3,3,1};
    const int mPad [8] = {2,2,2,1,1,0,1,0};
    const int mNC  [8] = {1,4,4,3,3,3,2,1};   // ceil(Cin*k*k / 32)
    const int mOff [8] = {0,1,5,9,12,15,18,20};

    _Float16* panel  = (_Float16*)d_ws;                      // 21 chunks * 512 halves
    short*    offtab = (short*)((char*)d_ws + 24576);        // 21 chunks * 512 shorts
    float*    wsum   = (float*)((char*)d_ws + 49152);        // 8 layers * 4
    float*    F      = (float*)((char*)d_ws + 65536);

    // Workspace carving (floats, 256B aligned)
    size_t off = 0;
    auto alloc = [&](size_t n) { float* p = F + off; off += (n + 63) & ~((size_t)63); return p; };
    const size_t full1 = (size_t)B * 1 * 480 * 640;
    const size_t full4 = (size_t)B * 4 * 480 * 640;
    const size_t half4 = (size_t)B * 4 * 240 * 320;
    const size_t quar4 = (size_t)B * 4 * 120 * 160;
    const size_t eig4  = (size_t)B * 4 * 60 * 80;
    const size_t o4    = (size_t)B * 4 * 478 * 638;

    float* c0  = alloc(full1);
    float* x1a = alloc(full4); float* c1a = alloc(full4);
    float* x1b = alloc(full4); float* c1b = alloc(full4);
    float* hx1 = alloc(half4); float* hc1 = alloc(half4);
    float* hx2a= alloc(half4); float* hc2a= alloc(half4);
    float* hx2b= alloc(half4); float* hc2b= alloc(half4);
    float* qx2 = alloc(quar4); float* qc2 = alloc(quar4);
    float* qx3 = alloc(quar4); float* qc3 = alloc(quar4);
    float* ex3 = alloc(eig4);  float* ec3 = alloc(eig4);
    float* ex4 = alloc(eig4);  float* ec4 = alloc(eig4);
    float* qx4 = alloc(quar4); float* qc4 = alloc(quar4);
    float* qx34= alloc(quar4); float* qc34= alloc(quar4);
    float* xoA = alloc(o4);    float* coA = alloc(o4);
    // reuse dead buffers
    float* hx34 = hx2a; float* hc34 = hc2a;      // half, x2a dead
    float* hx23 = hx1;  float* hc23 = hc1;       // half, x1ds dead
    float* x23  = x1b;  float* c23  = c1b;       // full, x1b dead after conv3
    float* xoB  = x1b;  float* coB  = c1b;       // o4 <= full4, x23 dead after w6
    float* x7   = c0;                            // o1 <= full1, c0 dead

    // 1) pack weights (softplus) into WMMA B panels + offset tables + wsums
    for (int l = 0; l < 8; ++l)
        pack_weights_kernel<<<mNC[l], 32, 0, stream>>>(
            Wt[l], panel + (size_t)mOff[l] * 512, offtab + (size_t)mOff[l] * 512,
            wsum + 4 * l, mCout[l], mCin[l], mK[l]);

    // 2) c0 = S > 0.01
    thresh_kernel<<<(int)((full1 + 255) / 256), 256, 0, stream>>>(S, c0, (int)full1);

    auto conv = [&](const float* xa, const float* ca, int chA,
                    const float* xb, const float* cb,
                    float* xo, float* co, int l, int Hin, int Win) {
        int ksz = mK[l], pad = mPad[l];
        int Hout = Hin + 2 * pad - ksz + 1;
        int Wout = Win + 2 * pad - ksz + 1;
        dim3 g((Wout + 15) / 16, Hout, B);
        nconv_wmma_kernel<<<g, 32, 0, stream>>>(
            xa, ca, chA, xb, cb, xo, co,
            panel + (size_t)mOff[l] * 512, offtab + (size_t)mOff[l] * 512,
            mNC[l], wsum + 4 * l, Bs[l],
            mCin[l], Hin, Win, ksz, pad, Hout, Wout, mCout[l]);
    };
    auto pool = [&](const float* x, const float* c, float* xo, float* co,
                    int Hin, int Win) {
        int h2 = Hin / 2, w2 = Win / 2;
        int total = B * 4 * h2 * w2;
        pool2_kernel<<<(total + 255) / 256, 256, 0, stream>>>(
            x, c, xo, co, 4, h2, w2, Hin, Win, total);
    };
    auto up2 = [&](const float* in, float* out, int hi, int wi) {
        int total = B * 4 * 2 * hi * 2 * wi;
        up2_kernel<<<(total + 255) / 256, 256, 0, stream>>>(in, out, 4, hi, wi, total);
    };

    // encoder, full res
    conv(S,   c0,  1, nullptr, nullptr, x1a, c1a, 0, 480, 640);   // w1
    conv(x1a, c1a, 4, nullptr, nullptr, x1b, c1b, 1, 480, 640);   // w2
    conv(x1b, c1b, 4, nullptr, nullptr, x1a, c1a, 2, 480, 640);   // w3 -> x1,c1
    // 1/2
    pool(x1a, c1a, hx1, hc1, 480, 640);
    conv(hx1,  hc1,  4, nullptr, nullptr, hx2a, hc2a, 1, 240, 320); // w2
    conv(hx2a, hc2a, 4, nullptr, nullptr, hx2b, hc2b, 2, 240, 320); // w3 -> x2_ds
    // 1/4
    pool(hx2b, hc2b, qx2, qc2, 240, 320);
    conv(qx2, qc2, 4, nullptr, nullptr, qx3, qc3, 1, 120, 160);     // w2 -> x3_ds
    // 1/8
    pool(qx3, qc3, ex3, ec3, 120, 160);
    conv(ex3, ec3, 4, nullptr, nullptr, ex4, ec4, 1, 60, 80);       // w2 -> x4_ds
    // decoder
    up2(ex4, qx4, 60, 80); up2(ec4, qc4, 60, 80);
    conv(qx3, qc3, 4, qx4, qc4, qx34, qc34, 3, 120, 160);           // w4, cat(x3_ds,x4)
    up2(qx34, hx34, 120, 160); up2(qc34, hc34, 120, 160);
    conv(hx2b, hc2b, 4, hx34, hc34, hx23, hc23, 4, 240, 320);       // w5, cat(x2_ds,x34)
    up2(hx23, x23, 240, 320); up2(hc23, c23, 240, 320);
    conv(x23, c23, 4, x1a, c1a, xoA, coA, 5, 480, 640);             // w6 pad0 -> 478x638
    conv(xoA, coA, 4, nullptr, nullptr, xoB, coB, 6, 478, 638);     // w65
    conv(xoB, coB, 4, nullptr, nullptr, x7, nullptr, 7, 478, 638);  // w7 1x1, 1ch

    // adaptive average resize 478x638 -> 480x640
    int totalOut = B * H * W;
    resize_kernel<<<(totalOut + 255) / 256, 256, 0, stream>>>(
        x7, (float*)d_out, 478, 638, H, W, totalOut);
}
